// MetaNeuralCVModel_lotka_51651276702410
// MI455X (gfx1250) — matrix operand order repaired and checked by reading, not compiled, not run on hardware
//
#include <hip/hip_runtime.h>

typedef __attribute__((ext_vector_type(16))) _Float16 v16h;
typedef __attribute__((ext_vector_type(8)))  _Float16 v8h;
typedef __attribute__((ext_vector_type(8)))  float    v8f;

#define H       256   // hidden width (H1 == H2 == 256)
#define DIN     8
#define NT      8     // samples per workgroup chunk
#define NCOLS   9     // 1 forward column + 8 tangent columns
#define THREADS 256   // 8 wave32s

// Branch-free tanh. gfx1250 has V_TANH_F32; fall back to (e-1)/(e+1) with
// v_exp_f32 + v_rcp_f32 (arg clamped so exp2 never overflows -> no inf/inf).
__device__ __forceinline__ float fast_tanhf(float x) {
#if __has_builtin(__builtin_amdgcn_tanhf)
  return __builtin_amdgcn_tanhf(x);
#else
  float xc = fminf(fmaxf(x, -44.0f), 44.0f);
  float e  = __builtin_amdgcn_exp2f(xc * 2.8853900817779268f); // 2*log2(e)
  return (e - 1.0f) * __builtin_amdgcn_rcpf(e + 1.0f);
#endif
}

__global__ __launch_bounds__(THREADS)
void mlp_jactrace_kernel(
    const float* __restrict__ xb,  const float* __restrict__ sc,
    const float* __restrict__ W0,  const float* __restrict__ b0,
    const float* __restrict__ W1,  const float* __restrict__ b1,
    const float* __restrict__ W2,  const float* __restrict__ b2p,
    const float* __restrict__ W3,  const float* __restrict__ b3,
    const float* __restrict__ cp,  float* __restrict__ out, int B)
{
  // B-matrix columns per sample, f16, column-major (k contiguous): 36 KB
  __shared__ __align__(16) _Float16 Bls[NT * NCOLS * H];
  // shared zero column for the 7 padding columns of each 16-wide N tile
  __shared__ __align__(16) _Float16 zcol[H];
  // per-sample per-column reduction accumulators (w2-weighted column sums)
  __shared__ float colsum[NT * 16];

  const int tid  = threadIdx.x;
  const int lane = tid & 31;
  const int wid  = tid >> 5;
  const int hf   = (lane >> 4) & 1;   // lane half (0: lanes 0-15, 1: lanes 16-31)
  const int r    = lane & 15;         // column index within tile / row index of A

  // thread tid owns row k = tid of layer 0
  float w0r[DIN], b0k;
#pragma unroll
  for (int d = 0; d < DIN; ++d) w0r[d] = W0[tid * DIN + d];
  b0k = b0[tid];

  zcol[tid] = (_Float16)0.f;          // tid < 256 == H
  __syncthreads();

  const int nChunks = (B + NT - 1) / NT;
  for (int chunk = blockIdx.x; chunk < nChunks; chunk += gridDim.x) {
    const int s0 = chunk * NT;

    // ---------- phase 1: zero colsum, build [y0 | (1-y0^2) * W0] in LDS ----------
    if (tid < NT * 16) colsum[tid] = 0.f;
    for (int s = 0; s < NT; ++s) {
      const int gs = s0 + s;
      float xv[DIN];
#pragma unroll
      for (int d = 0; d < DIN; ++d) xv[d] = (gs < B) ? xb[gs * DIN + d] : 0.f;
      float z0 = b0k;
#pragma unroll
      for (int d = 0; d < DIN; ++d) z0 += w0r[d] * xv[d];
      const float y0 = fast_tanhf(z0);
      const float d0 = 1.f - y0 * y0;
      Bls[(s * NCOLS) * H + tid] = (_Float16)y0;
#pragma unroll
      for (int d = 0; d < DIN; ++d)
        Bls[(s * NCOLS + 1 + d) * H + tid] = (_Float16)(d0 * w0r[d]);
    }
    __syncthreads();

    // ---------- phase 2: Z1 = W1 @ [y0 | T0] via WMMA + fused epilogue ----------
    for (int mt = wid; mt < 16; mt += 8) {
      const int m0 = mt * 16;

      // A strip: rows m0..m0+15 of W1, all K=256, as 8 f16 fragments (64 VGPRs).
      // A layout (16-bit, 16x32): lane r holds M=r; element e -> K = 8*hf+e (e<8)
      // or 16+8*hf+(e-8) (e>=8).
      v16h astrip[8];
      const float* w1row = W1 + (size_t)(m0 + r) * H;
#pragma unroll
      for (int kk = 0; kk < 8; ++kk) {
        const int kbase = kk * 32 + hf * 8;
        const float4* pa = (const float4*)(w1row + kbase);
        const float4* pb = (const float4*)(w1row + kbase + 16);
        float4 f0 = pa[0], f1 = pa[1], f2 = pb[0], f3 = pb[1];
        v16h a;
        a[0] = (_Float16)f0.x; a[1] = (_Float16)f0.y; a[2] = (_Float16)f0.z; a[3] = (_Float16)f0.w;
        a[4] = (_Float16)f1.x; a[5] = (_Float16)f1.y; a[6] = (_Float16)f1.z; a[7] = (_Float16)f1.w;
        a[8]  = (_Float16)f2.x; a[9]  = (_Float16)f2.y; a[10] = (_Float16)f2.z; a[11] = (_Float16)f2.w;
        a[12] = (_Float16)f3.x; a[13] = (_Float16)f3.y; a[14] = (_Float16)f3.z; a[15] = (_Float16)f3.w;
        astrip[kk] = a;
      }
      // per-lane bias / w2 for the C rows this lane owns: M = p + 8*hf
      float b1v[8], w2v[8];
#pragma unroll
      for (int p = 0; p < 8; ++p) {
        const int m = m0 + p + 8 * hf;
        b1v[p] = b1[m];
        w2v[p] = W2[m];
      }

      for (int s = 0; s < NT; ++s) {
        // B layout (16-bit, 32x16): lane r holds column N=r; element e -> K = 8*hf + e (e<8)
        // or 16+8*hf+(e-8) (e>=8)
        const _Float16* colbase = (r < NCOLS) ? &Bls[(s * NCOLS + r) * H] : zcol;

        // preload all 8 B fragments (16 back-to-back ds_load_b128), then issue
        // WMMAs as two independent accumulation chains.
        v16h bfr[8];
#pragma unroll
        for (int kk = 0; kk < 8; ++kk) {
          const _Float16* bp = colbase + kk * 32 + hf * 8;
          v8h lo = *(const v8h*)(bp);
          v8h hi = *(const v8h*)(bp + 16);
          bfr[kk] = __builtin_shufflevector(lo, hi,
              0, 1, 2, 3, 4, 5, 6, 7, 8, 9, 10, 11, 12, 13, 14, 15);
        }
        v8f acc0 = {0.f, 0.f, 0.f, 0.f, 0.f, 0.f, 0.f, 0.f};
        v8f acc1 = {0.f, 0.f, 0.f, 0.f, 0.f, 0.f, 0.f, 0.f};
#pragma unroll
        for (int kk = 0; kk < 8; kk += 2) {
          acc0 = __builtin_amdgcn_wmma_f32_16x16x32_f16(
              false, astrip[kk],     false, bfr[kk],     (short)0, acc0, false, false);
          acc1 = __builtin_amdgcn_wmma_f32_16x16x32_f16(
              false, astrip[kk + 1], false, bfr[kk + 1], (short)0, acc1, false, false);
        }
        v8f acc = acc0 + acc1;

        // epilogue: column 0 -> y1 = tanh(z1+b1); tangent cols j -> d1 * z1_j
        // then w2-weighted sum over this tile's 16 rows; d1 broadcast from the
        // column-0 lane of the same half (lane & 16). Branch-free.
        float partial = 0.f;
#pragma unroll
        for (int p = 0; p < 8; ++p) {
          const float z   = acc[p] + b1v[p];
          const float y1  = fast_tanhf(z);
          const float d1  = 1.f - y1 * y1;
          const float d1b = __shfl(d1, lane & 16, 32);
          const float t0  = w2v[p] * y1;           // forward column
          const float tj  = w2v[p] * d1b * acc[p]; // tangent columns (no bias)
          partial += (r == 0) ? t0 : tj;
        }
        if (r < NCOLS) atomicAdd(&colsum[s * 16 + r], partial);
      }
    }
    __syncthreads();

    // ---------- phase 3: finalize per sample ----------
    if (tid < NT) {
      const int s = tid, gs = s0 + s;
      if (gs < B) {
        const float y2pre = colsum[s * 16 + 0] + b2p[0];
        const float y2 = fast_tanhf(y2pre);
        const float d2 = 1.f - y2 * y2;
        float dotWs = 0.f, dotbs = 0.f, div = 0.f;
#pragma unroll
        for (int i = 0; i < DIN; ++i) {
          const float sci = sc[gs * DIN + i];
          dotWs += W3[i] * sci;
          dotbs += b3[i] * sci;
          div   += W3[i] * colsum[s * 16 + 1 + i];
        }
        out[gs] = cp[0] + d2 * div + y2 * dotWs + dotbs;
      }
    }
    __syncthreads();
  }
}

extern "C" void kernel_launch(void* const* d_in, const int* in_sizes, int n_in,
                              void* d_out, int out_size, void* d_ws, size_t ws_size,
                              hipStream_t stream) {
  (void)n_in; (void)d_ws; (void)ws_size; (void)out_size;
  const float* xb  = (const float*)d_in[0];
  const float* sc  = (const float*)d_in[1];
  const float* W0  = (const float*)d_in[2];
  const float* b0  = (const float*)d_in[3];
  const float* W1  = (const float*)d_in[4];
  const float* b1  = (const float*)d_in[5];
  const float* W2  = (const float*)d_in[6];
  const float* b2  = (const float*)d_in[7];
  const float* W3  = (const float*)d_in[8];
  const float* b3  = (const float*)d_in[9];
  const float* c   = (const float*)d_in[10];
  float* out = (float*)d_out;

  const int B = in_sizes[0] / DIN;
  const int nChunks = (B + NT - 1) / NT;
  const int grid = nChunks < 2048 ? nChunks : 2048;
  mlp_jactrace_kernel<<<grid, THREADS, 0, stream>>>(
      xb, sc, W0, b0, W1, b1, W2, b2, W3, b3, c, out, B);
}
